// Flow_7748121002234
// MI455X (gfx1250) — matrix operand order repaired and checked by reading, not compiled, not run on hardware
//
#include <hip/hip_runtime.h>
#include <hip/hip_bf16.h>

typedef _Float16 half_t;
typedef __attribute__((ext_vector_type(16))) _Float16 v16h;
typedef __attribute__((ext_vector_type(8)))  float    v8f;

#define THREADS 256
#define TILE_P  128

// ---------------------------------------------------------------------------
// Pack pointwise weights [CIN, COUT] f32 -> per-lane WMMA B-fragment layout,
// f16, K padded to KCH*32, N padded to NT*16.
// half index = (((nt*KCH + kc)*32) + lane)*16 + e ;  k = kc*32 + (lane>>4)*16 + e
// ---------------------------------------------------------------------------
__global__ void pack_w_kernel(const float* __restrict__ pw, half_t* __restrict__ out,
                              int CIN, int COUT, int KCH, int NT) {
  int total = NT * KCH * 512;
  int idx = blockIdx.x * THREADS + threadIdx.x;
  if (idx >= total) return;
  int e    = idx & 15;
  int lane = (idx >> 4) & 31;
  int rest = idx >> 9;
  int kc   = rest % KCH;
  int nt   = rest / KCH;
  int n = nt * 16 + (lane & 15);
  int k = kc * 32 + (lane >> 4) * 16 + e;
  out[idx] = (k < CIN && n < COUT) ? (half_t)pw[(size_t)k * COUT + n] : (half_t)0.f;
}

// ---------------------------------------------------------------------------
// Conv2DTranspose(2ch, 4x4, stride 2, SAME) * 2.0  (lax.conv_transpose pads 2,2
// on the x2-dilated input, kernel HWIO not flipped)
// ---------------------------------------------------------------------------
__global__ void upsample_kernel(const float* __restrict__ fin, const float* __restrict__ k44,
                                float* __restrict__ fout, int Hin, int Win) {
  int Ho = Hin * 2, Wo = Win * 2;
  int P = 4 * Ho * Wo;
  int p = blockIdx.x * THREADS + threadIdx.x;
  if (p >= P) return;
  int HWo = Ho * Wo;
  int b = p / HWo; int r = p - b * HWo; int y = r / Wo; int x = r - y * Wo;
  float s0 = 0.f, s1 = 0.f;
  for (int ky = 0; ky < 4; ++ky) {
    int ty = y - 2 + ky;
    if (ty < 0 || (ty & 1)) continue;
    int yi = ty >> 1; if (yi >= Hin) continue;
    for (int kx = 0; kx < 4; ++kx) {
      int tx = x - 2 + kx;
      if (tx < 0 || (tx & 1)) continue;
      int xi = tx >> 1; if (xi >= Win) continue;
      const float* iv = fin + ((size_t)(b * Hin * Win + yi * Win + xi)) * 2;
      const float* kv = k44 + (ky * 4 + kx) * 4;     // [i*2 + o]
      s0 += iv[0] * kv[0] + iv[1] * kv[2];
      s1 += iv[0] * kv[1] + iv[1] * kv[3];
    }
  }
  fout[(size_t)p * 2 + 0] = 2.f * s0;
  fout[(size_t)p * 2 + 1] = 2.f * s1;
}

// ---------------------------------------------------------------------------
// dense_image_warp(img, -flow): bilinear sample at (y+flow0, x+flow1), clamped
// ---------------------------------------------------------------------------
__global__ void warp_kernel(const float* __restrict__ img, const float* __restrict__ flow,
                            float* __restrict__ wc, int H, int W) {
  int P = 4 * H * W;
  int p = blockIdx.x * THREADS + threadIdx.x;
  if (p >= P) return;
  int HW = H * W;
  int b = p / HW; int r = p - b * HW; int y = r / W; int x = r - y * W;
  float qy = fminf(fmaxf((float)y + flow[(size_t)p * 2 + 0], 0.f), (float)(H - 1));
  float qx = fminf(fmaxf((float)x + flow[(size_t)p * 2 + 1], 0.f), (float)(W - 1));
  int y0 = (int)floorf(qy); y0 = y0 < 0 ? 0 : (y0 > H - 2 ? H - 2 : y0);
  int x0 = (int)floorf(qx); x0 = x0 < 0 ? 0 : (x0 > W - 2 ? W - 2 : x0);
  float ay = qy - (float)y0, ax = qx - (float)x0;
  const float* i00 = img + ((size_t)(b * HW + y0 * W + x0)) * 64;
  const float* i01 = i00 + 64;
  const float* i10 = i00 + (size_t)W * 64;
  const float* i11 = i10 + 64;
  float* o = wc + (size_t)p * 64;
#pragma unroll 8
  for (int c = 0; c < 64; ++c) {
    float top = i00[c] * (1.f - ax) + i01[c] * ax;
    float bot = i10[c] * (1.f - ax) + i11[c] * ax;
    o[c] = top * (1.f - ay) + bot * ay;
  }
}

// ---------------------------------------------------------------------------
// correlation (81 disps, mean over 64ch) + concat [corr|wc|prev|(fu)] -> f16
// ---------------------------------------------------------------------------
__global__ void corr_concat_kernel(const float* __restrict__ prev, const float* __restrict__ wc,
                                   const float* __restrict__ fu, half_t* __restrict__ xcat,
                                   int H, int W, int CIN) {
  int P = 4 * H * W;
  int p = blockIdx.x * THREADS + threadIdx.x;
  if (p >= P) return;
  int HW = H * W;
  int b = p / HW; int r = p - b * HW; int y = r / W; int x = r - y * W;
  const float4* pa = (const float4*)(prev + (size_t)p * 64);
  half_t* xr = xcat + (size_t)p * CIN;
  for (int dy = 0; dy < 9; ++dy) {
    int yy = y + dy - 4;
    for (int dx = 0; dx < 9; ++dx) {
      int xx = x + dx - 4;
      float s = 0.f;
      if (yy >= 0 && yy < H && xx >= 0 && xx < W) {
        const float4* pb = (const float4*)(wc + ((size_t)(b * HW + yy * W + xx)) * 64);
#pragma unroll
        for (int i = 0; i < 16; ++i) {
          float4 a4 = pa[i], b4 = pb[i];
          s += a4.x * b4.x + a4.y * b4.y + a4.z * b4.z + a4.w * b4.w;
        }
      }
      xr[dy * 9 + dx] = (half_t)(s * 0.015625f);   // mean over 64 channels
    }
  }
  const float* wr = wc + (size_t)p * 64;
  const float* pr = prev + (size_t)p * 64;
#pragma unroll 8
  for (int c = 0; c < 64; ++c) {
    xr[81 + c]  = (half_t)wr[c];
    xr[145 + c] = (half_t)pr[c];
  }
  if (fu) {
    xr[209] = (half_t)fu[(size_t)p * 2 + 0];
    xr[210] = (half_t)fu[(size_t)p * 2 + 1];
  }
}

// ---------------------------------------------------------------------------
// Fused SeparableConv2D stage:
//   phase 1: 3x3 depthwise (SAME, zero pad) over CIN channels -> f16 tile in LDS
//   phase 2: pointwise GEMM via v_wmma_f32_16x16x32_f16, K padded to CINP
//            (compile-time); per K-chunk: one clause of NT B-fragment loads,
//            one wait, then an NT-deep WMMA chain -> loads of the next chunk
//            overlap the matrix pipe.
//   epilogue: +bias, optional inference-BN + ReLU; f16 activations out,
//             or f32 2-channel flow out (FINAL).
// One workgroup = 8 wave32 = 128-pixel tile; each wave owns 16 pixels x NT*16.
// ---------------------------------------------------------------------------
template <int NT, int CINP, bool BN, bool FINAL>
__global__ __launch_bounds__(THREADS)
void sepconv_kernel(const half_t* __restrict__ xin, int CIN,
                    const float* __restrict__ dw,
                    const half_t* __restrict__ wB,
                    const float* __restrict__ bias,
                    const float* __restrict__ gamma,
                    const float* __restrict__ beta,
                    half_t* __restrict__ yout, float* __restrict__ flout,
                    int P, int H, int W) {
  constexpr int KCH = CINP / 32;
  __shared__ half_t At[TILE_P * CINP];
  const int tile0 = blockIdx.x * TILE_P;
  const int tid = threadIdx.x;
  const int HW = H * W;

  // ---- phase 1: depthwise conv into LDS (zero-pad OOB pixels & channels) ----
  const int total = TILE_P * CINP;
  for (int idx = tid; idx < total; idx += THREADS) {
    int pix = idx / CINP, c = idx - pix * CINP;
    int g = tile0 + pix;
    float acc = 0.f;
    if (g < P && c < CIN) {
      int b = g / HW; int r = g - b * HW; int y = r / W; int x = r - y * W;
#pragma unroll
      for (int ky = 0; ky < 3; ++ky) {
        int yy = y + ky - 1;
        if (yy < 0 || yy >= H) continue;
#pragma unroll
        for (int kx = 0; kx < 3; ++kx) {
          int xx = x + kx - 1;
          if (xx < 0 || xx >= W) continue;
          acc += (float)xin[(size_t)(b * HW + yy * W + xx) * CIN + c] *
                 dw[(ky * 3 + kx) * CIN + c];
        }
      }
    }
    At[pix * CINP + c] = (half_t)acc;
  }
  __syncthreads();

  // ---- phase 2: WMMA GEMM  (A from LDS tile, B pre-packed fragments) ----
  const int lane = tid & 31;
  const int wave = tid >> 5;
  const int rowBase = wave * 16;
  const int mrow = lane & 15;
  const int h = lane >> 4;

  v8f zero = {};
  v8f acc[NT];
#pragma unroll
  for (int t = 0; t < NT; ++t) acc[t] = zero;

  const v16h* wBv = (const v16h*)wB;
#pragma unroll
  for (int kc = 0; kc < KCH; ++kc) {
    // A fragment: 16x32 f16, contiguous K order matching packed B: K = h*16+e
    v16h a;
    const half_t* arow = &At[(rowBase + mrow) * CINP + kc * 32];
#pragma unroll
    for (int e = 0; e < 16; ++e) {
      int kk = h * 16 + e;
      a[e] = arow[kk];
    }
    // bulk-load all NT B fragments for this K-chunk (single clause, one wait)
    v16h bf[NT];
#pragma unroll
    for (int t = 0; t < NT; ++t)
      bf[t] = wBv[((t * KCH + kc) << 5) + lane];
#pragma unroll
    for (int t = 0; t < NT; ++t)
      acc[t] = __builtin_amdgcn_wmma_f32_16x16x32_f16(
          false, a, false, bf[t], (short)0, acc[t], false, false);
  }

  // ---- epilogue: C/D layout lane(n = lane&15), VGPR e -> row e + 8*h ----
#pragma unroll
  for (int t = 0; t < NT; ++t) {
    int n = t * 16 + mrow;
#pragma unroll
    for (int e = 0; e < 8; ++e) {
      int m = e + 8 * h;
      int g = tile0 + rowBase + m;
      if (g >= P) continue;
      if (FINAL) {
        if (n < 2) flout[(size_t)g * 2 + n] = acc[t][e] + bias[n];
      } else {
        float val = acc[t][e] + bias[n];
        if (BN) val = fmaxf(0.f, val * gamma[n] + beta[n]);
        yout[(size_t)g * 128 + n] = (half_t)val;
      }
    }
  }
}

// ---------------------------------------------------------------------------
// Host-side orchestration
// d_in: current_0, prev_0, ..., current_4, prev_4 (0..9), then params pytree
// leaves (sorted keys): fe0..fe4 (13 arrays each: b, blk0{b,beta,dw,gamma,pw},
// blk1{...}, dw, pw), then up1..up4.
// ---------------------------------------------------------------------------
extern "C" void kernel_launch(void* const* d_in, const int* in_sizes, int n_in,
                              void* d_out, int out_size, void* d_ws, size_t ws_size,
                              hipStream_t stream) {
  (void)in_sizes; (void)n_in; (void)out_size; (void)ws_size;
  const int Hs[5] = {128, 64, 32, 16, 8};
  const int Ws[5] = {256, 128, 64, 32, 16};
  size_t Pa[5];
  for (int l = 0; l < 5; ++l) Pa[l] = (size_t)4 * Hs[l] * Ws[l];

  // d_out layout (floats): flow_0..flow_4, flow_1_up..flow_4_up
  float* outf = (float*)d_out;
  size_t flowOff[5], upOff[5], acc = 0;
  for (int l = 0; l < 5; ++l) { flowOff[l] = acc; acc += Pa[l] * 2; }
  for (int l = 1; l <= 4; ++l) { upOff[l] = acc; acc += Pa[l - 1] * 2; }

  // workspace carve-up
  char* wsb = (char*)d_ws;
  size_t wo = 0;
  auto wsAlloc = [&](size_t bytes) -> void* {
    void* p = wsb + wo;
    wo = (wo + bytes + 255) & ~(size_t)255;
    return p;
  };
  half_t* xcat = (half_t*)wsAlloc(Pa[0] * 211 * sizeof(half_t));
  half_t* yA   = (half_t*)wsAlloc(Pa[0] * 128 * sizeof(half_t));
  half_t* yB   = (half_t*)wsAlloc(Pa[0] * 128 * sizeof(half_t));
  float*  wcb  = (float*)wsAlloc(Pa[0] * 64 * sizeof(float));
  half_t *wpk1[5], *wpk2[5], *wpk3[5];
  for (int l = 0; l < 5; ++l) {
    wpk1[l] = (half_t*)wsAlloc((size_t)8 * 7 * 512 * sizeof(half_t)); // 224x128
    wpk2[l] = (half_t*)wsAlloc((size_t)8 * 4 * 512 * sizeof(half_t)); // 128x128
    wpk3[l] = (half_t*)wsAlloc((size_t)1 * 4 * 512 * sizeof(half_t)); // 128x16
  }

  dim3 blk(THREADS);

  // pack all pointwise weights to f16 WMMA fragments (recomputed every call)
  for (int l = 0; l < 5; ++l) {
    int fb = 10 + 13 * l;
    int CIN = (l == 4) ? 209 : 211;
    pack_w_kernel<<<dim3((8 * 7 * 512 + THREADS - 1) / THREADS), blk, 0, stream>>>(
        (const float*)d_in[fb + 5], wpk1[l], CIN, 128, 7, 8);
    pack_w_kernel<<<dim3((8 * 4 * 512 + THREADS - 1) / THREADS), blk, 0, stream>>>(
        (const float*)d_in[fb + 10], wpk2[l], 128, 128, 4, 8);
    pack_w_kernel<<<dim3((1 * 4 * 512 + THREADS - 1) / THREADS), blk, 0, stream>>>(
        (const float*)d_in[fb + 12], wpk3[l], 128, 2, 4, 1);
  }

  for (int lvl = 4; lvl >= 0; --lvl) {
    const int H = Hs[lvl], W = Ws[lvl];
    const int P = (int)Pa[lvl];
    const int CIN = (lvl == 4) ? 209 : 211;
    const float* cur = (const float*)d_in[2 * lvl];
    const float* prv = (const float*)d_in[2 * lvl + 1];
    const int fb = 10 + 13 * lvl;
    dim3 gP((P + THREADS - 1) / THREADS);

    if (lvl == 4) {
      corr_concat_kernel<<<gP, blk, 0, stream>>>(prv, cur, nullptr, xcat, H, W, CIN);
    } else {
      const int lu = lvl + 1;
      const float* upk = (const float*)d_in[74 + lu];     // up1 -> 75 ... up4 -> 78
      float* fu = outf + upOff[lu];
      upsample_kernel<<<gP, blk, 0, stream>>>(outf + flowOff[lu], upk, fu, Hs[lu], Ws[lu]);
      warp_kernel<<<gP, blk, 0, stream>>>(cur, fu, wcb, H, W);
      corr_concat_kernel<<<gP, blk, 0, stream>>>(prv, wcb, fu, xcat, H, W, CIN);
    }

    dim3 gT((P + TILE_P - 1) / TILE_P);
    sepconv_kernel<8, 224, true, false><<<gT, blk, 0, stream>>>(
        xcat, CIN,
        (const float*)d_in[fb + 3], wpk1[lvl], (const float*)d_in[fb + 1],
        (const float*)d_in[fb + 4], (const float*)d_in[fb + 2],
        yA, nullptr, P, H, W);
    sepconv_kernel<8, 128, true, false><<<gT, blk, 0, stream>>>(
        yA, 128,
        (const float*)d_in[fb + 8], wpk2[lvl], (const float*)d_in[fb + 6],
        (const float*)d_in[fb + 9], (const float*)d_in[fb + 7],
        yB, nullptr, P, H, W);
    sepconv_kernel<1, 128, false, true><<<gT, blk, 0, stream>>>(
        yB, 128,
        (const float*)d_in[fb + 11], wpk3[lvl], (const float*)d_in[fb + 0],
        nullptr, nullptr,
        nullptr, outf + flowOff[lvl], P, H, W);
  }
}
